// FCGRU_61787399520725
// MI455X (gfx1250) — compile-verified, matrix operand order checked by reading, not run on hardware
//
#include <hip/hip_runtime.h>
#include <hip/hip_bf16.h>

typedef __bf16 v8bf  __attribute__((ext_vector_type(8)));
typedef __bf16 v16bf __attribute__((ext_vector_type(16)));
typedef float  v8f   __attribute__((ext_vector_type(8)));

#define NROWS   16384
#define IN_DIM  512
#define HID     1024
#define DIM2    512
#define OUT_DIM 128
#define TILE_M  32

#define WMMA_BF16(A, Bm, C) \
    __builtin_amdgcn_wmma_f32_16x16x32_bf16(false, (A), false, (Bm), (short)0, (C), false, false)

// ---------------- WMMA fragment load (ISA 7.12.2 16-bit layout) ---------------
// lane L supplies row (L&15); halves 0..7 are K = k0 + (L>>4)*8,
// halves 8..15 are K = k0 + 16 + (L>>4)*8  -> two 16-byte loads.
__device__ __forceinline__ v16bf ld_frag_bf16(const __bf16* base, int ld,
                                              int r0, int k0, int lane) {
    const int r  = r0 + (lane & 15);
    const int kk = k0 + ((lane >> 4) << 3);
    const __bf16* p = base + r * ld + kk;
    v8bf lo = *(const v8bf*)(p);
    v8bf hi = *(const v8bf*)(p + 16);
    return __builtin_shufflevector(lo, hi, 0,1,2,3,4,5,6,7,8,9,10,11,12,13,14,15);
}

__device__ __forceinline__ float sigmoid_f(float x) {
    return 1.0f / (1.0f + __expf(-x));
}

// ------- tile GEMM stage: Out = act(Ain @ W^T + bias), both row slices per B --
// Ain: [TILE_M][K] bf16 (LDS), W: [N][K] bf16 (global), Out: [TILE_M][N]
template<bool RELU, typename OutT>
__device__ __forceinline__ void gemm_stage(const __bf16* Ain, int K,
                                           const __bf16* __restrict__ W,
                                           const float* __restrict__ bias,
                                           OutT* Out, int N, int wave, int lane) {
    const int NT = N >> 4;
    for (int ct = wave; ct < NT; ct += 8) {
        const int c = ct << 4;
        v8f acc0 = {}, acc1 = {};
        for (int k = 0; k < K; k += 32) {
            v16bf a0 = ld_frag_bf16(Ain, K, 0,  k, lane);
            v16bf a1 = ld_frag_bf16(Ain, K, 16, k, lane);
            v16bf b  = ld_frag_bf16(W,   K, c,  k, lane);   // load B once,
            acc0 = WMMA_BF16(a0, b, acc0);                  // use it twice
            acc1 = WMMA_BF16(a1, b, acc1);
        }
        const int n  = lane & 15;
        const int mb = (lane >> 4) << 3;
        const float bv = bias[c + n];
#pragma unroll
        for (int i = 0; i < 8; ++i) {
            float v0 = acc0[i] + bv;
            float v1 = acc1[i] + bv;
            if (RELU) { v0 = fmaxf(v0, 0.0f); v1 = fmaxf(v1, 0.0f); }
            Out[(mb + i) * N + (c + n)]        = (OutT)v0;
            Out[(16 + mb + i) * N + (c + n)]   = (OutT)v1;
        }
    }
}

// ---------------- f32 -> bf16 weight conversion ------------------------------
__global__ void cvt_f32_bf16_kernel(const float* __restrict__ src,
                                    __bf16* __restrict__ dst, int n) {
    int i = blockIdx.x * blockDim.x + threadIdx.x;
    if (i < n) dst[i] = (__bf16)src[i];
}

// ---------------- fused megakernel -------------------------------------------
__global__ void __launch_bounds__(256)
fcgru_fused_kernel(const float* __restrict__ input,
                   const float* __restrict__ hn,
                   const int*   __restrict__ cult,
                   const __bf16* __restrict__ W1b,  const float* __restrict__ b1,
                   const __bf16* __restrict__ W2b,  const float* __restrict__ b2,
                   const __bf16* __restrict__ Wihb, const __bf16* __restrict__ Whhb,
                   const float* __restrict__ b_ih,  const float* __restrict__ b_hh,
                   const __bf16* __restrict__ W3b,  const float* __restrict__ b3,
                   const __bf16* __restrict__ Whb,  const float* __restrict__ bh,
                   float* __restrict__ out_params,  float* __restrict__ out_h) {
    extern __shared__ unsigned char smem[];
    // LDS unions (192 KB total):
    __bf16* Xb  = (__bf16*)(smem);               // [32][512]  (stage 1 in)
    __bf16* A1b = (__bf16*)(smem + 32 * 1024);   // [32][512]  (stage 1 out)
    __bf16* Gb  = (__bf16*)(smem);               // [32][1024] (reuse Xb+A1b)
    __bf16* A2b = (__bf16*)(smem + 64 * 1024);   // [32][1024] (stage 2 out)
    float*  G2f = (float*)(smem + 64 * 1024);    // [32][512]  (reuse A2b)
    __bf16* Hb  = (__bf16*)(smem + 128 * 1024);  // [32][1024]

    const int tid  = threadIdx.x;
    const int lane = tid & 31;
    const int wave = tid >> 5;
    const int row_base = blockIdx.x * TILE_M;

    // ---- load + convert X and H tiles into LDS (bf16) ----
    {
        const float4* ip = (const float4*)(input + (size_t)row_base * IN_DIM);
        for (int i = tid; i < TILE_M * IN_DIM / 4; i += 256) {
            float4 v = ip[i]; int o = i * 4;
            Xb[o+0] = (__bf16)v.x; Xb[o+1] = (__bf16)v.y;
            Xb[o+2] = (__bf16)v.z; Xb[o+3] = (__bf16)v.w;
        }
        const float4* hp = (const float4*)(hn + (size_t)row_base * HID);
        for (int i = tid; i < TILE_M * HID / 4; i += 256) {
            float4 v = hp[i]; int o = i * 4;
            Hb[o+0] = (__bf16)v.x; Hb[o+1] = (__bf16)v.y;
            Hb[o+2] = (__bf16)v.z; Hb[o+3] = (__bf16)v.w;
        }
    }
    __syncthreads();

    // ---- stage 1: A1 = X @ W1^T + b1 ----
    gemm_stage<false, __bf16>(Xb, IN_DIM, W1b, b1, A1b, DIM2, wave, lane);
    __syncthreads();
    // ---- stage 2: A2 = A1 @ W2^T + b2 ----
    gemm_stage<false, __bf16>(A1b, DIM2, W2b, b2, A2b, HID, wave, lane);
    __syncthreads();

    // ---- stage 3: fused GRU cell; 6 gate slices x 2 row slices per col tile --
    {
        const int NT = HID >> 4;  // 64 column tiles
        for (int ct = wave; ct < NT; ct += 8) {
            const int c = ct << 4;
            v8f air0 = {}, aiz0 = {}, ain0 = {}, ahr0 = {}, ahz0 = {}, ahn0 = {};
            v8f air1 = {}, aiz1 = {}, ain1 = {}, ahr1 = {}, ahz1 = {}, ahn1 = {};
            for (int k = 0; k < HID; k += 32) {
                v16bf a20 = ld_frag_bf16(A2b, HID, 0,  k, lane);
                v16bf a21 = ld_frag_bf16(A2b, HID, 16, k, lane);
                v16bf b;
                b = ld_frag_bf16(Wihb, HID, c, k, lane);
                air0 = WMMA_BF16(a20, b, air0); air1 = WMMA_BF16(a21, b, air1);
                b = ld_frag_bf16(Wihb, HID, HID + c, k, lane);
                aiz0 = WMMA_BF16(a20, b, aiz0); aiz1 = WMMA_BF16(a21, b, aiz1);
                b = ld_frag_bf16(Wihb, HID, 2 * HID + c, k, lane);
                ain0 = WMMA_BF16(a20, b, ain0); ain1 = WMMA_BF16(a21, b, ain1);
                v16bf h0 = ld_frag_bf16(Hb, HID, 0,  k, lane);
                v16bf h1 = ld_frag_bf16(Hb, HID, 16, k, lane);
                b = ld_frag_bf16(Whhb, HID, c, k, lane);
                ahr0 = WMMA_BF16(h0, b, ahr0); ahr1 = WMMA_BF16(h1, b, ahr1);
                b = ld_frag_bf16(Whhb, HID, HID + c, k, lane);
                ahz0 = WMMA_BF16(h0, b, ahz0); ahz1 = WMMA_BF16(h1, b, ahz1);
                b = ld_frag_bf16(Whhb, HID, 2 * HID + c, k, lane);
                ahn0 = WMMA_BF16(h0, b, ahn0); ahn1 = WMMA_BF16(h1, b, ahn1);
            }
            const int n   = lane & 15;
            const int mb  = (lane >> 4) << 3;
            const int col = c + n;
            const float bir = b_ih[col], biz = b_ih[HID + col], bin = b_ih[2 * HID + col];
            const float bhr = b_hh[col], bhz = b_hh[HID + col], bhn = b_hh[2 * HID + col];

            auto combine = [&](int r0, const v8f& air, const v8f& aiz, const v8f& ain,
                               const v8f& ahr, const v8f& ahz, const v8f& ahn) {
#pragma unroll
                for (int i = 0; i < 8; ++i) {
                    const int lrow = r0 + mb + i;
                    const int grow = row_base + lrow;
                    float r  = sigmoid_f(air[i] + bir + ahr[i] + bhr);
                    float z  = sigmoid_f(aiz[i] + biz + ahz[i] + bhz);
                    float nn = tanhf(ain[i] + bin + r * (ahn[i] + bhn));
                    float hp = hn[(size_t)grow * HID + col];   // f32 h_prev
                    float hnew = (1.0f - z) * nn + z * hp;
                    out_h[(size_t)grow * HID + col] = hnew;
                    Gb[lrow * HID + col] = (__bf16)fmaxf(hnew, 0.0f);
                }
            };
            combine(0,  air0, aiz0, ain0, ahr0, ahz0, ahn0);
            combine(16, air1, aiz1, ain1, ahr1, ahz1, ahn1);
        }
    }
    __syncthreads();

    // ---- stage 4: G2 = relu(G @ W3^T + b3), f32 into LDS ----
    gemm_stage<true, float>(Gb, HID, W3b, b3, G2f, DIM2, wave, lane);
    __syncthreads();

    // ---- stage 5: per-cultivar head (VALU dot products) ----
    for (int p = tid; p < TILE_M * OUT_DIM; p += 256) {
        const int lrow = p & (TILE_M - 1);
        const int o    = p >> 5;           // TILE_M == 32
        const int grow = row_base + lrow;
        const int idx  = cult[grow];
        const v8bf* wp = (const v8bf*)(Whb + ((size_t)(idx * OUT_DIM + o) << 9));
        const float* gp = G2f + lrow * DIM2;
        float acc = 0.0f;
        for (int d = 0; d < DIM2 / 8; ++d) {
            v8bf w = wp[d];
            const float* g = gp + d * 8;
#pragma unroll
            for (int j = 0; j < 8; ++j) acc += g[j] * (float)w[j];
        }
        out_params[(size_t)grow * OUT_DIM + o] = acc + bh[idx * OUT_DIM + o];
    }
}

// ---------------- host-side launch -------------------------------------------
extern "C" void kernel_launch(void* const* d_in, const int* in_sizes, int n_in,
                              void* d_out, int out_size, void* d_ws, size_t ws_size,
                              hipStream_t stream) {
    const float* input = (const float*)d_in[0];
    const float* hn    = (const float*)d_in[1];
    const int*   cult  = (const int*)d_in[2];
    const float* W1    = (const float*)d_in[3];
    const float* b1    = (const float*)d_in[4];
    const float* W2    = (const float*)d_in[5];
    const float* b2    = (const float*)d_in[6];
    const float* W_ih  = (const float*)d_in[7];
    const float* W_hh  = (const float*)d_in[8];
    const float* b_ih  = (const float*)d_in[9];
    const float* b_hh  = (const float*)d_in[10];
    const float* W3    = (const float*)d_in[11];
    const float* b3    = (const float*)d_in[12];
    const float* Wh    = (const float*)d_in[13];
    const float* bh    = (const float*)d_in[14];

    float* out_params = (float*)d_out;
    float* out_h      = out_params + (size_t)NROWS * OUT_DIM;

    // bf16 weight staging in workspace
    __bf16* wsb = (__bf16*)d_ws;
    const size_t nW1  = (size_t)DIM2 * IN_DIM;       // 262144
    const size_t nW2  = (size_t)HID * DIM2;          // 524288
    const size_t nWih = (size_t)3 * HID * HID;       // 3145728
    const size_t nWhh = nWih;
    const size_t nW3  = (size_t)DIM2 * HID;          // 524288
    const size_t nWh  = (size_t)32 * OUT_DIM * DIM2; // 2097152
    __bf16* W1b  = wsb;
    __bf16* W2b  = W1b  + nW1;
    __bf16* Wihb = W2b  + nW2;
    __bf16* Whhb = Wihb + nWih;
    __bf16* W3b  = Whhb + nWhh;
    __bf16* Whb  = W3b  + nW3;

    auto cvt = [&](const float* s, __bf16* d, size_t n) {
        cvt_f32_bf16_kernel<<<(unsigned)((n + 255) / 256), 256, 0, stream>>>(s, d, (int)n);
    };
    cvt(W1,  W1b,  nW1);
    cvt(W2,  W2b,  nW2);
    cvt(W_ih, Wihb, nWih);
    cvt(W_hh, Whhb, nWhh);
    cvt(W3,  W3b,  nW3);
    cvt(Wh,  Whb,  nWh);

    const int blocks = NROWS / TILE_M;   // 512
    const size_t lds_bytes = 192 * 1024; // Xb/A1b|Gb + A2b|G2f + Hb
    fcgru_fused_kernel<<<blocks, 256, lds_bytes, stream>>>(
        input, hn, cult,
        W1b, b1, W2b, b2, Wihb, Whhb, b_ih, b_hh, W3b, b3, Whb, bh,
        out_params, out_h);
}
